// SwinV23DMHSA_14748917695096
// MI455X (gfx1250) — matrix operand order repaired and checked by reading, not compiled, not run on hardware
//
#include <hip/hip_runtime.h>
#include <hip/hip_bf16.h>
#include <math.h>

typedef __attribute__((ext_vector_type(16))) _Float16 v16h;
typedef __attribute__((ext_vector_type(8)))  _Float16 half8;
typedef __attribute__((ext_vector_type(8)))  float    v8f;

#define DIMC   512
#define HEADS  16
#define SEQ    256
#define BATCH  128
#define HD     32          // head dim
#define NQKV   1536
#define CPBH   512
#define TBL    1575        // 7*15*15

// ---------- fragment helpers ----------
__device__ __forceinline__ v16h frag_from_f16(const _Float16* lo, const _Float16* hi) {
    half8 a = *(const half8*)lo;
    half8 b = *(const half8*)hi;
    return __builtin_shufflevector(a, b, 0,1,2,3,4,5,6,7,8,9,10,11,12,13,14,15);
}

__device__ __forceinline__ v8f wmma_f16(v16h a, v16h b, v8f c) {
    return __builtin_amdgcn_wmma_f32_16x16x32_f16(false, a, false, b, (short)0, c, false, false);
}

// ---------- kernel 0: bulk f32 -> f16 conversion ----------
__global__ void cvt_f16_kernel(const float* __restrict__ src, _Float16* __restrict__ dst, int n) {
    int i = (blockIdx.x * blockDim.x + threadIdx.x) * 8;
    if (i >= n) return;
    float4 a = *(const float4*)(src + i);
    float4 b = *(const float4*)(src + i + 4);
    _Float16 o[8] = {(_Float16)a.x, (_Float16)a.y, (_Float16)a.z, (_Float16)a.w,
                     (_Float16)b.x, (_Float16)b.y, (_Float16)b.z, (_Float16)b.w};
    *(half8*)(dst + i) = *(half8*)o;
}

// ---------- kernel 1: CPB MLP -> bias_table[1575][16] ----------
__global__ void cpb_mlp_kernel(const float* __restrict__ coords,  // [1575,3]
                               const float* __restrict__ w1,      // [512,3]
                               const float* __restrict__ b1,      // [512]
                               const float* __restrict__ w2,      // [16,512]
                               float* __restrict__ table)         // [1575,16]
{
    int idx = blockIdx.x * blockDim.x + threadIdx.x;
    if (idx >= TBL * HEADS) return;
    int row = idx >> 4, h = idx & 15;
    float x0 = coords[row * 3 + 0], x1 = coords[row * 3 + 1], x2 = coords[row * 3 + 2];
    float acc = 0.f;
    for (int j = 0; j < CPBH; ++j) {
        float hj = fmaf(w1[j * 3 + 0], x0, fmaf(w1[j * 3 + 1], x1, fmaf(w1[j * 3 + 2], x2, b1[j])));
        hj = fmaxf(hj, 0.f);
        acc = fmaf(w2[h * CPBH + j], hj, acc);
    }
    table[row * HEADS + h] = acc;
}

// ---------- kernel 2: gather + 16*sigmoid -> bias[H][S][S] ----------
__global__ void bias_gather_kernel(const float* __restrict__ table,  // [1575,16]
                                   const int* __restrict__ relidx,   // [S*S]
                                   float* __restrict__ bias)         // [H,S,S]
{
    int idx = blockIdx.x * blockDim.x + threadIdx.x;
    if (idx >= HEADS * SEQ * SEQ) return;
    int h = idx >> 16;
    int ij = idx & 0xFFFF;
    float v = table[relidx[ij] * HEADS + h];
    bias[idx] = 16.f / (1.f + __expf(-v));
}

// ---------- kernel 3: QKV GEMM (all-f16 operands) + cosine-norm epilogue ----------
// wave tile 32(M) x 64(N): 8 accumulators, 2 A-frags + 4 B-frags, 8 WMMA per k-step.
// 64-col span = two heads of one q/k/v section.
// out: qh[bh][s][d] f16 (scaled+normed), kh[bh][s][d] f16 (normed), vt[bh][d][s] f16
__global__ __launch_bounds__(256)
void qkv_gemm_kernel(const _Float16* __restrict__ X,   // [32768,512] f16
                     const _Float16* __restrict__ W,   // [1536,512] f16 (== B^T)
                     const float* __restrict__ bqkv,   // [1536]
                     const float* __restrict__ lscale, // [16]
                     _Float16* __restrict__ qh,
                     _Float16* __restrict__ kh,
                     _Float16* __restrict__ vt)
{
    int wid  = blockIdx.x * 8 + (threadIdx.x >> 5);
    int lane = threadIdx.x & 31;
    int lh = lane >> 4, ln = lane & 15;
    int nt = wid % (NQKV / 64);
    int mt = wid / (NQKV / 64);
    int m0 = mt * 32, n0 = nt * 64;

    const _Float16* ar[2];
    ar[0] = X + (size_t)(m0 + ln) * DIMC;
    ar[1] = X + (size_t)(m0 + 16 + ln) * DIMC;
    const _Float16* br[4];
#pragma unroll
    for (int ct = 0; ct < 4; ++ct) br[ct] = W + (size_t)(n0 + ct * 16 + ln) * DIMC;

    v8f acc[2][4];
#pragma unroll
    for (int i = 0; i < 2; ++i)
#pragma unroll
        for (int j = 0; j < 4; ++j) acc[i][j] = (v8f){};

#pragma unroll 2
    for (int ks = 0; ks < DIMC; ks += 32) {
        __builtin_prefetch(ar[0] + ks + 32, 0, 1);
        __builtin_prefetch(br[0] + ks + 32, 0, 1);
        v16h a0 = frag_from_f16(ar[0] + ks + lh * 8, ar[0] + ks + 16 + lh * 8);
        v16h a1 = frag_from_f16(ar[1] + ks + lh * 8, ar[1] + ks + 16 + lh * 8);
#pragma unroll
        for (int ct = 0; ct < 4; ++ct) {
            v16h b = frag_from_f16(br[ct] + ks + lh * 8, br[ct] + ks + 16 + lh * 8);
            acc[0][ct] = wmma_f16(a0, b, acc[0][ct]);
            acc[1][ct] = wmma_f16(a1, b, acc[1][ct]);
        }
    }

    // epilogue: n0 multiple of 64 -> one qkv section, heads h0 and h0+1
    int qkv_idx = n0 >> 9;            // 0=q 1=k 2=v
    int h0      = (n0 >> 5) & 15;
    float sc0 = __expf(fminf(lscale[h0], 4.60517019f));
    float sc1 = __expf(fminf(lscale[h0 + 1], 4.60517019f));
    float bq[4];
#pragma unroll
    for (int ct = 0; ct < 4; ++ct) bq[ct] = bqkv[n0 + ct * 16 + ln];

#pragma unroll
    for (int rt = 0; rt < 2; ++rt) {
#pragma unroll
        for (int r = 0; r < 8; ++r) {
            float v0 = acc[rt][0][r] + bq[0];
            float v1 = acc[rt][1][r] + bq[1];
            float v2 = acc[rt][2][r] + bq[2];
            float v3 = acc[rt][3][r] + bq[3];
            int m  = m0 + rt * 16 + lh * 8 + r;   // token index
            int bi = m >> 8, s = m & 255;
            size_t bh0 = (size_t)bi * HEADS + h0;
            size_t bh1 = bh0 + 1;
            if (qkv_idx == 2) {
                vt[bh0 * (HD * SEQ) + (size_t)ln * SEQ + s]        = (_Float16)v0;
                vt[bh0 * (HD * SEQ) + (size_t)(16 + ln) * SEQ + s] = (_Float16)v1;
                vt[bh1 * (HD * SEQ) + (size_t)ln * SEQ + s]        = (_Float16)v2;
                vt[bh1 * (HD * SEQ) + (size_t)(16 + ln) * SEQ + s] = (_Float16)v3;
            } else {
                // L2 norm over d=32 per head: 16-lane half reduction (xor<=8 stays in-half)
                float ssa = v0 * v0 + v1 * v1;
                float ssb = v2 * v2 + v3 * v3;
                ssa += __shfl_xor(ssa, 1); ssb += __shfl_xor(ssb, 1);
                ssa += __shfl_xor(ssa, 2); ssb += __shfl_xor(ssb, 2);
                ssa += __shfl_xor(ssa, 4); ssb += __shfl_xor(ssb, 4);
                ssa += __shfl_xor(ssa, 8); ssb += __shfl_xor(ssb, 8);
                float inva = 1.0f / fmaxf(sqrtf(ssa), 1e-12f);
                float invb = 1.0f / fmaxf(sqrtf(ssb), 1e-12f);
                if (qkv_idx == 0) { inva *= sc0; invb *= sc1; }
                _Float16* dst = (qkv_idx == 0) ? qh : kh;
                dst[bh0 * (SEQ * HD) + (size_t)s * HD + ln]      = (_Float16)(v0 * inva);
                dst[bh0 * (SEQ * HD) + (size_t)s * HD + 16 + ln] = (_Float16)(v1 * inva);
                dst[bh1 * (SEQ * HD) + (size_t)s * HD + ln]      = (_Float16)(v2 * invb);
                dst[bh1 * (SEQ * HD) + (size_t)s * HD + 16 + ln] = (_Float16)(v3 * invb);
            }
        }
    }
}

// ---------- kernel 4: attention (scores+bias -> softmax -> P@V) ----------
// grid: B*H*4 workgroups of 128 threads (4 waves); each WG does 64 query rows.
__global__ __launch_bounds__(128)
void attn_kernel(const _Float16* __restrict__ qh,
                 const _Float16* __restrict__ kh,
                 const _Float16* __restrict__ vt,
                 const float* __restrict__ biasw,   // [H,S,S]
                 _Float16* __restrict__ ctx)        // [B*S, DIM] f16
{
    __shared__ float sc[64 * 256];   // 64KB: scores, then packed f16 P in-place

    int wg = blockIdx.x;
    int q4 = wg & 3;
    int h  = (wg >> 2) & 15;
    int b  = wg >> 6;
    size_t bh = (size_t)b * HEADS + h;
    int tid = threadIdx.x;
    int w = tid >> 5, lane = tid & 31, lh = lane >> 4, ln = lane & 15;
    int i_base = q4 * 64;
    int i0 = i_base + w * 16;

    // --- phase 1: S1 = qn @ kn^T + bias  (K = d = 32 -> single WMMA per tile)
    const _Float16* qrow = qh + bh * (SEQ * HD) + (size_t)(i0 + ln) * HD;
    v16h aq = frag_from_f16(qrow + lh * 8, qrow + 16 + lh * 8);
    const float* brow = biasw + (size_t)h * (SEQ * SEQ);

    for (int jt = 0; jt < 16; ++jt) {
        const _Float16* krow = kh + bh * (SEQ * HD) + (size_t)(jt * 16 + ln) * HD;
        v16h bk = frag_from_f16(krow + lh * 8, krow + 16 + lh * 8);
        v8f c = {};
        c = wmma_f16(aq, bk, c);
#pragma unroll
        for (int r = 0; r < 8; ++r) {
            int il = w * 16 + lh * 8 + r;
            int j  = jt * 16 + ln;
            sc[il * 256 + j] = c[r] + brow[(size_t)(i_base + il) * SEQ + j];
        }
    }
    __syncthreads();

    // --- phase 2: row softmax, pack to f16 in place (row stride stays 1024B)
    if (tid < 64) {
        float* row = sc + tid * 256;
        float mx = -3.0e38f;
        for (int j = 0; j < 256; ++j) mx = fmaxf(mx, row[j]);
        _Float16* prow = (_Float16*)row;
        float sum = 0.f;
        for (int j = 0; j < 256; ++j) {
            float e = __expf(row[j] - mx);   // write at 2j clobbers row[j/2]: already read
            sum += e;
            prow[j] = (_Float16)e;
        }
        float is = 1.0f / sum;
        for (int j = 0; j < 256; ++j) prow[j] = (_Float16)((float)prow[j] * is);
    }
    __syncthreads();

    // --- phase 3: ctx = P @ V  (M=16/wave, N=32, K=256; A from LDS, B from vt)
    v8f c0 = {}, c1 = {};
    const _Float16* prow = (const _Float16*)(sc + (size_t)(w * 16 + ln) * 256);
    const _Float16* v0r = vt + bh * (HD * SEQ) + (size_t)ln * SEQ;
    const _Float16* v1r = vt + bh * (HD * SEQ) + (size_t)(16 + ln) * SEQ;
#pragma unroll
    for (int ksi = 0; ksi < 8; ++ksi) {
        int kk = ksi * 32;
        v16h ap = frag_from_f16(prow + kk + lh * 8, prow + kk + 16 + lh * 8);
        v16h b0 = frag_from_f16(v0r + kk + lh * 8, v0r + kk + 16 + lh * 8);
        v16h b1 = frag_from_f16(v1r + kk + lh * 8, v1r + kk + 16 + lh * 8);
        c0 = wmma_f16(ap, b0, c0);
        c1 = wmma_f16(ap, b1, c1);
    }
#pragma unroll
    for (int r = 0; r < 8; ++r) {
        int s = i_base + w * 16 + lh * 8 + r;
        size_t base = ((size_t)b * SEQ + s) * DIMC + h * HD;
        ctx[base + ln]      = (_Float16)c0[r];
        ctx[base + 16 + ln] = (_Float16)c1[r];
    }
}

// ---------- kernel 5: output projection (all-f16 operands) ----------
// wave tile 32(M) x 64(N)
__global__ __launch_bounds__(256)
void proj_gemm_kernel(const _Float16* __restrict__ ctx,  // [32768,512] f16
                      const _Float16* __restrict__ W,    // [512,512] f16 (== B^T)
                      const float* __restrict__ bproj,   // [512]
                      float* __restrict__ out)           // [32768,512]
{
    int wid  = blockIdx.x * 8 + (threadIdx.x >> 5);
    int lane = threadIdx.x & 31;
    int lh = lane >> 4, ln = lane & 15;
    int nt = wid & 7;           // 512/64
    int mt = wid >> 3;
    int m0 = mt * 32, n0 = nt * 64;

    const _Float16* ar[2];
    ar[0] = ctx + (size_t)(m0 + ln) * DIMC;
    ar[1] = ctx + (size_t)(m0 + 16 + ln) * DIMC;
    const _Float16* br[4];
#pragma unroll
    for (int ct = 0; ct < 4; ++ct) br[ct] = W + (size_t)(n0 + ct * 16 + ln) * DIMC;

    v8f acc[2][4];
#pragma unroll
    for (int i = 0; i < 2; ++i)
#pragma unroll
        for (int j = 0; j < 4; ++j) acc[i][j] = (v8f){};

#pragma unroll 2
    for (int ks = 0; ks < DIMC; ks += 32) {
        __builtin_prefetch(ar[0] + ks + 32, 0, 1);
        __builtin_prefetch(br[0] + ks + 32, 0, 1);
        v16h a0 = frag_from_f16(ar[0] + ks + lh * 8, ar[0] + ks + 16 + lh * 8);
        v16h a1 = frag_from_f16(ar[1] + ks + lh * 8, ar[1] + ks + 16 + lh * 8);
#pragma unroll
        for (int ct = 0; ct < 4; ++ct) {
            v16h b = frag_from_f16(br[ct] + ks + lh * 8, br[ct] + ks + 16 + lh * 8);
            acc[0][ct] = wmma_f16(a0, b, acc[0][ct]);
            acc[1][ct] = wmma_f16(a1, b, acc[1][ct]);
        }
    }

    float bp[4];
#pragma unroll
    for (int ct = 0; ct < 4; ++ct) bp[ct] = bproj[n0 + ct * 16 + ln];
#pragma unroll
    for (int rt = 0; rt < 2; ++rt) {
#pragma unroll
        for (int r = 0; r < 8; ++r) {
            int m = m0 + rt * 16 + lh * 8 + r;
#pragma unroll
            for (int ct = 0; ct < 4; ++ct)
                out[(size_t)m * DIMC + n0 + ct * 16 + ln] = acc[rt][ct][r] + bp[ct];
        }
    }
}

// ---------- launch ----------
extern "C" void kernel_launch(void* const* d_in, const int* in_sizes, int n_in,
                              void* d_out, int out_size, void* d_ws, size_t ws_size,
                              hipStream_t stream) {
    const float* hidden = (const float*)d_in[0];
    const float* Wqkv   = (const float*)d_in[1];
    const float* bqkv   = (const float*)d_in[2];
    const float* Wproj  = (const float*)d_in[3];
    const float* bproj  = (const float*)d_in[4];
    const float* lscale = (const float*)d_in[5];
    const float* cpbw1  = (const float*)d_in[6];
    const float* cpbb1  = (const float*)d_in[7];
    const float* cpbw2  = (const float*)d_in[8];
    const float* coords = (const float*)d_in[9];
    const int*   relidx = (const int*)d_in[10];
    float* out = (float*)d_out;

    // workspace layout
    char* ws = (char*)d_ws;
    size_t off = 0;
    const size_t sz_q   = (size_t)BATCH * HEADS * SEQ * HD * sizeof(_Float16); // 32MB
    const size_t sz_ctx = (size_t)BATCH * SEQ * DIMC * sizeof(_Float16);       // 32MB
    _Float16* qh  = (_Float16*)(ws + off); off += sz_q;
    _Float16* kh  = (_Float16*)(ws + off); off += sz_q;
    _Float16* vt  = (_Float16*)(ws + off); off += sz_q;
    _Float16* ctx = (_Float16*)(ws + off); off += sz_ctx;
    float* biasw  = (float*)(ws + off);    off += (size_t)HEADS * SEQ * SEQ * sizeof(float);
    float* table  = (float*)(ws + off);    off += (size_t)TBL * HEADS * sizeof(float);
    off = (off + 255) & ~(size_t)255;
    _Float16* Xh  = (_Float16*)(ws + off); off += (size_t)BATCH * SEQ * DIMC * sizeof(_Float16);
    _Float16* Wqh = (_Float16*)(ws + off); off += (size_t)NQKV * DIMC * sizeof(_Float16);
    _Float16* Wph = (_Float16*)(ws + off); off += (size_t)DIMC * DIMC * sizeof(_Float16);

    // 0) one-pass f32 -> f16 of activations and weights
    {
        int nX = BATCH * SEQ * DIMC;          // 16,777,216
        int nWq = NQKV * DIMC;                // 786,432
        int nWp = DIMC * DIMC;                // 262,144
        cvt_f16_kernel<<<(nX / 8 + 255) / 256, 256, 0, stream>>>(hidden, Xh, nX);
        cvt_f16_kernel<<<(nWq / 8 + 255) / 256, 256, 0, stream>>>(Wqkv, Wqh, nWq);
        cvt_f16_kernel<<<(nWp / 8 + 255) / 256, 256, 0, stream>>>(Wproj, Wph, nWp);
    }
    // 1) CPB MLP table
    cpb_mlp_kernel<<<(TBL * HEADS + 255) / 256, 256, 0, stream>>>(coords, cpbw1, cpbb1, cpbw2, table);
    // 2) gather + sigmoid bias
    bias_gather_kernel<<<(HEADS * SEQ * SEQ) / 256, 256, 0, stream>>>(table, relidx, biasw);
    // 3) QKV GEMM + norm epilogue: 1024 m-tiles x 24 n-tiles (64 wide), 8 waves/block
    qkv_gemm_kernel<<<(1024 * 24) / 8, 256, 0, stream>>>(Xh, Wqh, bqkv, lscale, qh, kh, vt);
    // 4) attention: B*H*4 workgroups of 128 threads
    attn_kernel<<<BATCH * HEADS * 4, 128, 0, stream>>>(qh, kh, vt, biasw, ctx);
    // 5) projection: 1024 m-tiles x 8 n-tiles (64 wide), 8 waves/block
    proj_gemm_kernel<<<(1024 * 8) / 8, 256, 0, stream>>>(ctx, Wph, bproj, out);
}
